// StructuredReasoner_80642305950480
// MI455X (gfx1250) — matrix operand order repaired and checked by reading, not compiled, not run on hardware
//
#include <hip/hip_runtime.h>

typedef unsigned short ushort_t;
typedef __attribute__((ext_vector_type(16))) __bf16 bf16x16;
typedef __attribute__((ext_vector_type(8)))  float  floatx8;
typedef __attribute__((ext_vector_type(4)))  unsigned int uint4v;
typedef __attribute__((ext_vector_type(8)))  unsigned int uint8v;

constexpr int Bdim = 4, Sdim = 4096, Ddim = 2048;
constexpr int Rdim = 256, Edim = 8, Hdim = 512;
constexpr int Tdim = Bdim * Sdim;  // 16384

// ---------------------------------------------------------------- helpers
__device__ __forceinline__ unsigned short f2bf(float f) {
  unsigned int u = __float_as_uint(f);
  u += 0x7FFFu + ((u >> 16) & 1u);   // round-to-nearest-even
  return (unsigned short)(u >> 16);
}

// A fragment: A row-major [M x K] bf16. ISA layout: lane&15 = row,
// lane>>4 selects K halves {0..7,16..23} vs {8..15,24..31}.
__device__ __forceinline__ bf16x16 load_a_frag(const ushort_t* A, int lda,
                                               int m0, int k0, int lane) {
  const int m = m0 + (lane & 15);
  const int g = (lane >> 4) << 3;
  union { bf16x16 v; uint4 q[2]; } u;
  const ushort_t* p = A + (size_t)m * lda + k0;
  u.q[0] = *(const uint4*)(p + g);
  u.q[1] = *(const uint4*)(p + 16 + g);
  return u.v;
}

// B fragment: Bt stored as [N x K] row-major bf16 (pre-transposed weights).
// lane&15 = column n, lane>>4 selects K halves {0..7,16..23} vs {8..15,24..31}.
__device__ __forceinline__ bf16x16 load_b_frag(const ushort_t* Bt, int ldb,
                                               int n0, int k0, int lane) {
  const int n = n0 + (lane & 15);
  const int kb = k0 + ((lane >> 4) << 3);
  union { bf16x16 v; uint4 q[2]; } u;
  const ushort_t* p = Bt + (size_t)n * ldb + kb;
  u.q[0] = *(const uint4*)p;
  u.q[1] = *(const uint4*)(p + 16);
  return u.v;
}

__device__ __forceinline__ floatx8 wmma_bf16(bf16x16 a, bf16x16 b, floatx8 c) {
  return __builtin_amdgcn_wmma_f32_16x16x32_bf16(false, a, false, b,
                                                 (short)0, c, false, false);
}

// ---------------------------------------------------------------- converts
__global__ __launch_bounds__(256) void cvt_bf16_kernel(
    const float* __restrict__ in, ushort_t* __restrict__ out, size_t n4) {
  const size_t i = (size_t)blockIdx.x * 256 + threadIdx.x;
  if (i >= n4) return;
  const float4 v = *(const float4*)(in + i * 4);
  uint2 o;
  o.x = (unsigned)f2bf(v.x) | ((unsigned)f2bf(v.y) << 16);
  o.y = (unsigned)f2bf(v.z) | ((unsigned)f2bf(v.w) << 16);
  *(uint2*)(out + i * 4) = o;
}

// in: [batch][rows][cols] f32 -> out: [batch][cols][rows] bf16
__global__ __launch_bounds__(256) void tcvt_kernel(
    const float* __restrict__ in, ushort_t* __restrict__ out,
    int rows, int cols, int batch) {
  const size_t total = (size_t)rows * cols * batch;
  const size_t i = (size_t)blockIdx.x * 256 + threadIdx.x;
  if (i >= total) return;
  const size_t slice = (size_t)rows * cols;
  const size_t s = i / slice, rem = i % slice;
  const int c = (int)(rem / rows), r = (int)(rem % rows);
  out[s * slice + (size_t)c * rows + r] = f2bf(in[s * slice + (size_t)r * cols + c]);
}

// ---------------------------------------------------------------- big GEMM
// C[M,N] f32 (+ optional bf16 copy) = A[M,K]bf16 * Bt[N,K]bf16
// 128x64 tile per 256-thread WG; 8 waves as 4(m) x 2(n); 2x2 wmma tiles/wave.
__global__ __launch_bounds__(256) void gemm_bf16_kernel(
    const ushort_t* __restrict__ A, const ushort_t* __restrict__ Bt,
    float* __restrict__ C, ushort_t* __restrict__ Cb, int M, int N, int K) {
  const int lane = threadIdx.x & 31;
  const int w = threadIdx.x >> 5;
  const int m0 = blockIdx.y * 128 + (w >> 1) * 32;
  const int n0 = blockIdx.x * 64 + (w & 1) * 32;

  floatx8 acc[2][2] = {};
  for (int k0 = 0; k0 < K; k0 += 32) {
    const bf16x16 a0 = load_a_frag(A, K, m0, k0, lane);
    const bf16x16 a1 = load_a_frag(A, K, m0 + 16, k0, lane);
    const bf16x16 b0 = load_b_frag(Bt, K, n0, k0, lane);
    const bf16x16 b1 = load_b_frag(Bt, K, n0 + 16, k0, lane);
    acc[0][0] = wmma_bf16(a0, b0, acc[0][0]);
    acc[0][1] = wmma_bf16(a0, b1, acc[0][1]);
    acc[1][0] = wmma_bf16(a1, b0, acc[1][0]);
    acc[1][1] = wmma_bf16(a1, b1, acc[1][1]);
  }
  const int nn = n0 + (lane & 15);
  const int mb = m0 + ((lane >> 4) << 3);
  for (int i = 0; i < 2; ++i)
    for (int j = 0; j < 2; ++j)
#pragma unroll
      for (int v = 0; v < 8; ++v) {
        const size_t idx = (size_t)(mb + i * 16 + v) * N + (nn + j * 16);
        const float val = acc[i][j][v];
        C[idx] = val;
        if (Cb) Cb[idx] = f2bf(val);
      }
}

// ---------------------------------------------------------------- router
// one wave per token: coalesced z reads, shfl reduction, softmax+top2.
__global__ __launch_bounds__(256) void router_kernel(
    const float* __restrict__ z, const float* __restrict__ Wr,
    const float* __restrict__ br, float* __restrict__ combine) {
  const int lane = threadIdx.x & 31;
  const int t = blockIdx.x * 8 + (threadIdx.x >> 5);
  float lg[Edim];
#pragma unroll
  for (int e = 0; e < Edim; ++e) lg[e] = 0.f;
  const float* zrow = z + (size_t)t * Rdim;
  for (int r = lane; r < Rdim; r += 32) {
    const float zv = zrow[r];
#pragma unroll
    for (int e = 0; e < Edim; ++e) lg[e] += zv * Wr[r * Edim + e];
  }
#pragma unroll
  for (int e = 0; e < Edim; ++e)
    for (int off = 16; off > 0; off >>= 1) lg[e] += __shfl_xor(lg[e], off, 32);
  float mx = -1e30f;
#pragma unroll
  for (int e = 0; e < Edim; ++e) { lg[e] += br[e]; mx = fmaxf(mx, lg[e]); }
  float p[Edim], s = 0.f;
#pragma unroll
  for (int e = 0; e < Edim; ++e) { p[e] = expf(lg[e] - mx); s += p[e]; }
#pragma unroll
  for (int e = 0; e < Edim; ++e) p[e] /= s;
  int i1 = 0;
#pragma unroll
  for (int e = 1; e < Edim; ++e) if (p[e] > p[i1]) i1 = e;
  int i2 = (i1 == 0) ? 1 : 0;
#pragma unroll
  for (int e = 0; e < Edim; ++e) if (e != i1 && p[e] > p[i2]) i2 = e;
  const float sw = p[i1] + p[i2];
  if (lane < Edim) {
    const float cv = (lane == i1) ? p[i1] / sw : ((lane == i2) ? p[i2] / sw : 0.f);
    combine[(size_t)t * Edim + lane] = cv;
  }
}

// ---------------------------------------------------------------- fused MoE
// 32 tokens/WG. z tile DMA'd into LDS by the Tensor Data Mover (with HW
// padding giving the 528B conflict-free pitch); combine tile via async
// global->LDS loads. Loops all 8 experts with combine-weighted register
// accumulation, then does the 0.25/0.75 blend.
__global__ __launch_bounds__(256) void moe_kernel(
    const ushort_t* __restrict__ zb, const float* __restrict__ zf32,
    const float* __restrict__ combine,
    const ushort_t* __restrict__ W1bt, const float* __restrict__ bias1,
    const ushort_t* __restrict__ W2bt, const float* __restrict__ bias2,
    ushort_t* __restrict__ zfb) {
  __shared__ __align__(16) ushort_t zt[32 * 264];  // 32 x 256 (pitch 264 = 528B)
  __shared__ __align__(16) ushort_t ht[32 * 520];  // 32 x 512 (pitch 520)
  __shared__ float cmb[32 * Edim];

  const int tid = threadIdx.x;
  const int lane = tid & 31;
  const int w = tid >> 5;
  const int t0 = blockIdx.x * 32;

  // ---- TDM: DMA the 32-row x 512B z tile into LDS, padding 16B per 512B row.
  if (tid < 32) {
    const unsigned lds_base = (unsigned)(uintptr_t)(&zt[0]);
    const unsigned long long ga =
        (unsigned long long)(uintptr_t)(zb + (size_t)t0 * Rdim);
    uint4v g0;
    g0.x = 1u;                                       // count=1 (valid D#)
    g0.y = lds_base;                                 // lds_addr
    g0.z = (unsigned)(ga & 0xFFFFFFFFull);           // global_addr[31:0]
    g0.w = (unsigned)((ga >> 32) & 0x01FFFFFFull)    // global_addr[56:32]
           | (2u << 30);                             // type = 2 ("image")
    uint8v g1;
    g1[0] = (2u << 16)        // data_size = 4B elements
          | (1u << 20)        // pad_enable
          | (6u << 22)        // pad_interval: every 128 DWORDs (512B)
          | (3u << 25);       // pad_amount: 4 DWORDs (16B)
    g1[1] = (128u << 16);     // tensor_dim0 = 128 DWORDs per row
    g1[2] = (32u << 16);      // tensor_dim1 = 32 rows
    g1[3] = (128u << 16);     // tile_dim0 = 128 DWORDs
    g1[4] = 32u;              // tile_dim1 = 32 rows, tile_dim2 = 0
    g1[5] = 128u;             // tensor_dim0_stride = 128 DWORDs
    g1[6] = 0u;
    g1[7] = 0u;
    asm volatile("tensor_load_to_lds %0, %1" :: "s"(g0), "s"(g1) : "memory");
    __builtin_amdgcn_s_wait_tensorcnt(0);
  }
  // ---- async global->LDS for the 32x8 combine tile (1 B32 per thread).
  {
    const unsigned ldsc = (unsigned)(uintptr_t)(&cmb[tid]);
    const unsigned long long gac =
        (unsigned long long)(uintptr_t)(combine + (size_t)t0 * Edim + tid);
    asm volatile("global_load_async_to_lds_b32 %0, %1, off"
                 :: "v"(ldsc), "v"(gac) : "memory");
    asm volatile("s_wait_asynccnt 0" ::: "memory");
  }
  __syncthreads();

  floatx8 oacc[2][2] = {};  // persistent 32x(w-strip of 32) output accum

  for (int e = 0; e < Edim; ++e) {
    // ---- hidden = gelu(z * W1[e] + b1[e]) ; wave covers 64 cols of H
    const ushort_t* B1 = W1bt + (size_t)e * Hdim * Rdim;  // [H][R]
    __builtin_prefetch(B1 + ((size_t)tid << 9), 0, 1);    // pull slice toward L2
    for (int nt = 0; nt < 4; ++nt) {
      const int n0 = w * 64 + nt * 16;
      floatx8 hacc[2] = {};
      for (int k0 = 0; k0 < Rdim; k0 += 32) {
        const bf16x16 bb = load_b_frag(B1, Rdim, n0, k0, lane);
        const bf16x16 a0 = load_a_frag(zt, 264, 0, k0, lane);
        const bf16x16 a1 = load_a_frag(zt, 264, 16, k0, lane);
        hacc[0] = wmma_bf16(a0, bb, hacc[0]);
        hacc[1] = wmma_bf16(a1, bb, hacc[1]);
      }
      const int nn = n0 + (lane & 15);
      const float bv = bias1[e * Hdim + nn];
      const int mb = (lane >> 4) << 3;
      for (int i = 0; i < 2; ++i)
#pragma unroll
        for (int v = 0; v < 8; ++v) {
          const float x = hacc[i][v] + bv;
          const float g = 0.5f * x * (1.0f + erff(x * 0.70710678118654752440f));
          ht[(mb + i * 16 + v) * 520 + nn] = f2bf(g);
        }
    }
    __syncthreads();

    // ---- expert_out = hidden * W2[e] + b2[e] ; wave covers 32 cols of R
    const ushort_t* B2 = W2bt + (size_t)e * Rdim * Hdim;  // [R][H]
    __builtin_prefetch(B2 + ((size_t)tid << 9), 0, 1);
    floatx8 pacc[2][2] = {};
    const int n0o = w * 32;
    for (int k0 = 0; k0 < Hdim; k0 += 32) {
      const bf16x16 b0 = load_b_frag(B2, Hdim, n0o, k0, lane);
      const bf16x16 b1 = load_b_frag(B2, Hdim, n0o + 16, k0, lane);
      const bf16x16 a0 = load_a_frag(ht, 520, 0, k0, lane);
      const bf16x16 a1 = load_a_frag(ht, 520, 16, k0, lane);
      pacc[0][0] = wmma_bf16(a0, b0, pacc[0][0]);
      pacc[0][1] = wmma_bf16(a0, b1, pacc[0][1]);
      pacc[1][0] = wmma_bf16(a1, b0, pacc[1][0]);
      pacc[1][1] = wmma_bf16(a1, b1, pacc[1][1]);
    }
    {
      const int mb = (lane >> 4) << 3;
      for (int i = 0; i < 2; ++i)
#pragma unroll
        for (int v = 0; v < 8; ++v) {
          const int row = mb + i * 16 + v;
          const float cw = cmb[row * Edim + e];
#pragma unroll
          for (int j = 0; j < 2; ++j) {
            const int nc = n0o + j * 16 + (lane & 15);
            oacc[i][j][v] += cw * (pacc[i][j][v] + bias2[e * Rdim + nc]);
          }
        }
    }
    __syncthreads();
  }

  // ---- z_final = 0.25 * z_new + 0.75 * z  -> bf16 for the U reconstruction
  const int mb = (lane >> 4) << 3;
  for (int i = 0; i < 2; ++i)
    for (int j = 0; j < 2; ++j)
#pragma unroll
      for (int v = 0; v < 8; ++v) {
        const size_t row = (size_t)t0 + mb + i * 16 + v;
        const size_t col = (size_t)(w * 32 + j * 16 + (lane & 15));
        const float zf = 0.25f * oacc[i][j][v] + 0.75f * zf32[row * Rdim + col];
        zfb[row * Rdim + col] = f2bf(zf);
      }
}

// ---------------------------------------------------------------- halting
__global__ __launch_bounds__(256) void halt_kernel(
    const float* __restrict__ hnew, const float* __restrict__ Wh,
    const float* __restrict__ bh, float* __restrict__ phalt) {
  const int lane = threadIdx.x & 31;
  const int t = blockIdx.x * 8 + (threadIdx.x >> 5);
  float acc = 0.f;
  const float* row = hnew + (size_t)t * Ddim;
  for (int d = lane; d < Ddim; d += 32) acc += row[d] * Wh[d];
  for (int off = 16; off > 0; off >>= 1) acc += __shfl_xor(acc, off, 32);
  if (lane == 0) phalt[t] = 1.0f / (1.0f + expf(-(acc + bh[0])));
}

// ---------------------------------------------------------------- launch
extern "C" void kernel_launch(void* const* d_in, const int* in_sizes, int n_in,
                              void* d_out, int out_size, void* d_ws,
                              size_t ws_size, hipStream_t stream) {
  (void)in_sizes; (void)n_in; (void)out_size; (void)ws_size;
  const float* h  = (const float*)d_in[0];
  const float* V  = (const float*)d_in[1];
  const float* U  = (const float*)d_in[2];
  const float* Wr = (const float*)d_in[3];
  const float* br = (const float*)d_in[4];
  const float* W1 = (const float*)d_in[5];
  const float* b1 = (const float*)d_in[6];
  const float* W2 = (const float*)d_in[7];
  const float* b2 = (const float*)d_in[8];
  const float* Wh = (const float*)d_in[9];
  const float* bh = (const float*)d_in[10];

  char* ws = (char*)d_ws;
  size_t off = 0;
  auto alloc = [&](size_t bytes) {
    char* p = ws + off;
    off = (off + bytes + 255) & ~(size_t)255;
    return p;
  };
  ushort_t* hb   = (ushort_t*)alloc((size_t)Tdim * Ddim * 2);
  ushort_t* Vbt  = (ushort_t*)alloc((size_t)Rdim * Ddim * 2);
  ushort_t* Ubt  = (ushort_t*)alloc((size_t)Ddim * Rdim * 2);
  ushort_t* W1bt = (ushort_t*)alloc((size_t)Edim * Hdim * Rdim * 2);
  ushort_t* W2bt = (ushort_t*)alloc((size_t)Edim * Rdim * Hdim * 2);
  float*    z    = (float*)alloc((size_t)Tdim * Rdim * 4);
  ushort_t* zbuf = (ushort_t*)alloc((size_t)Tdim * Rdim * 2);
  float*    comb = (float*)alloc((size_t)Tdim * Edim * 4);
  ushort_t* zfb  = (ushort_t*)alloc((size_t)Tdim * Rdim * 2);

  float* hnew  = (float*)d_out;
  float* phalt = hnew + (size_t)Tdim * Ddim;

  cvt_bf16_kernel<<<(unsigned)(((size_t)Tdim * Ddim / 4) / 256), 256, 0, stream>>>(
      h, hb, (size_t)Tdim * Ddim / 4);
  tcvt_kernel<<<(Ddim * Rdim + 255) / 256, 256, 0, stream>>>(V, Vbt, Ddim, Rdim, 1);
  tcvt_kernel<<<(Rdim * Ddim + 255) / 256, 256, 0, stream>>>(U, Ubt, Rdim, Ddim, 1);
  tcvt_kernel<<<(Edim * Rdim * Hdim + 255) / 256, 256, 0, stream>>>(W1, W1bt, Rdim, Hdim, Edim);
  tcvt_kernel<<<(Edim * Hdim * Rdim + 255) / 256, 256, 0, stream>>>(W2, W2bt, Hdim, Rdim, Edim);

  gemm_bf16_kernel<<<dim3(Rdim / 64, Tdim / 128), 256, 0, stream>>>(
      hb, Vbt, z, zbuf, Tdim, Rdim, Ddim);
  router_kernel<<<Tdim / 8, 256, 0, stream>>>(z, Wr, br, comb);
  moe_kernel<<<Tdim / 32, 256, 0, stream>>>(zbuf, z, comb, W1bt, b1, W2bt, b2, zfb);
  gemm_bf16_kernel<<<dim3(Ddim / 64, Tdim / 128), 256, 0, stream>>>(
      zfb, Ubt, hnew, nullptr, Tdim, Ddim, Rdim);
  halt_kernel<<<Tdim / 8, 256, 0, stream>>>(hnew, Wh, bh, phalt);
}
